// LSTM_84095459656158
// MI455X (gfx1250) — compile-verified, matrix operand order checked by reading, not run on hardware
//
#include <hip/hip_runtime.h>

// ---------------------------------------------------------------------------
// LSTM on gfx1250: persistent kernel, LDS-resident bf16 weights, WMMA gates.
// T=512, B=64, I=512, H=1024, HX=1536, 4H=4096.
// ---------------------------------------------------------------------------

#define T_STEPS 512
#define BATCH   64
#define IN_SZ   512
#define HID     1024
#define HX      1536          // HID + IN_SZ
#define KTILES  48            // HX / 32
#define NWG     64            // HID / 16 hidden-columns per workgroup
#define WG_WELEMS 98304       // 4 gates * 16 cols * HX bf16 elems per workgroup
#define WG_WBYTES 196608      // bytes of packed weights per workgroup (fits 320KB LDS)

typedef __attribute__((ext_vector_type(16))) __bf16 v16bf;
typedef __attribute__((ext_vector_type(8)))  float  v8f;

// ---- helpers --------------------------------------------------------------

__device__ __forceinline__ unsigned short f2bf(float f) {
    unsigned u = __builtin_bit_cast(unsigned, f);
    u += 0x7fffu + ((u >> 16) & 1u);              // RNE
    return (unsigned short)(u >> 16);
}

__device__ __forceinline__ float sig_f(float x) {
    return 1.0f / (1.0f + __expf(-x));
}

__device__ __forceinline__ float tanh_f(float x) {
    float s = (x < 0.0f) ? -1.0f : 1.0f;
    float e = __expf(-2.0f * fabsf(x));
    return s * (1.0f - e) / (1.0f + e);
}

// ---- prep kernels ---------------------------------------------------------

// x [T,B,I] f32 -> bf16
__global__ void lstm_cvt_x(const float* __restrict__ x, unsigned short* __restrict__ xb) {
    int i = (blockIdx.x * blockDim.x + threadIdx.x) * 4;
    #pragma unroll
    for (int j = 0; j < 4; ++j) xb[i + j] = f2bf(x[i + j]);
}

// Pack Wf/Wi/Wc/Wo into per-workgroup fragment-major bf16 blocks:
// wpack[wg][kt][gate][lane][16] ; lane&15 = hidden col, lane>>4 = K half.
__global__ void lstm_pack_w(const float* __restrict__ Wf, const float* __restrict__ Wi,
                            const float* __restrict__ Wc, const float* __restrict__ Wo,
                            unsigned short* __restrict__ wpack) {
    int id = blockIdx.x * blockDim.x + threadIdx.x;       // NWG*KTILES*4*32 = 393216
    if (id >= NWG * KTILES * 4 * 32) return;
    int lane = id & 31;
    int gate = (id >> 5) & 3;
    int kt   = (id >> 7) % KTILES;
    int wg   = id / (32 * 4 * KTILES);
    const float* W = (gate == 0) ? Wf : (gate == 1) ? Wi : (gate == 2) ? Wc : Wo;
    int row = wg * 16 + (lane & 15);                       // hidden column this lane serves
    int kb  = kt * 32 + ((lane >> 4) << 4);                // K base for this lane half
    unsigned short* dst = wpack + (size_t)wg * WG_WELEMS
                        + (((size_t)(kt * 4 + gate) * 32 + lane) << 4);
    const float* src = W + (size_t)row * HX + kb;
    #pragma unroll
    for (int j = 0; j < 16; ++j) dst[j] = f2bf(src[j]);
}

// zero h0 buffer + grid-sync counter (re-run every launch: deterministic)
__global__ void lstm_init(unsigned short* __restrict__ h0, unsigned* __restrict__ cnt) {
    int i = blockIdx.x * blockDim.x + threadIdx.x;
    if (i < BATCH * HID) h0[i] = 0;
    if (i == 0) *cnt = 0u;
}

// ---- persistent LSTM kernel ----------------------------------------------
// grid = 64 workgroups (one per 16 hidden cols), 128 threads = 4 wave32.
// Wave w handles batch tile w (16 batches) and all 4 gate accumulators.

__global__ void __launch_bounds__(128, 1)
lstm_main(const unsigned short* __restrict__ wpack,
          const __bf16* __restrict__ xb,
          __bf16* __restrict__ hbuf0, __bf16* __restrict__ hbuf1,
          unsigned* __restrict__ cnt,
          const float* __restrict__ bf_, const float* __restrict__ bi_,
          const float* __restrict__ bc_, const float* __restrict__ bo_,
          float* __restrict__ out) {
    extern __shared__ __align__(16) char smemraw[];       // 196608 B dynamic LDS
    const int tid  = threadIdx.x;
    const int lane = tid & 31;
    const int mt   = tid >> 5;                             // batch tile 0..3
    const int wg   = blockIdx.x;
    const int n0   = wg * 16;

    // One-time: stage this block's packed weights in LDS (stay resident 512 steps)
    {
        const uint4* src = (const uint4*)(wpack + (size_t)wg * WG_WELEMS);
        uint4* dst = (uint4*)smemraw;
        for (int i = tid; i < WG_WBYTES / 16; i += 128) dst[i] = src[i];
    }
    __syncthreads();
    const __bf16* lw = (const __bf16*)smemraw;

    const int col = n0 + (lane & 15);
    const float biasF = bf_[col], biasI = bi_[col], biasC = bc_[col], biasO = bo_[col];

    v8f c = {0.f, 0.f, 0.f, 0.f, 0.f, 0.f, 0.f, 0.f};     // cell state, lives in VGPRs

    const int arow    = mt * 16 + (lane & 15);             // batch row this lane loads
    const int khalf16 = (lane >> 4) << 4;

    for (int t = 0; t < T_STEPS; ++t) {
        const __bf16* hr = (t & 1) ? hbuf1 : hbuf0;        // h_{t-1}
        __bf16*       hw = (t & 1) ? hbuf0 : hbuf1;        // h_t

        v8f accF = {0.f,0.f,0.f,0.f,0.f,0.f,0.f,0.f};
        v8f accI = accF, accC = accF, accO = accF;

        const __bf16* hrow = hr + (size_t)arow * HID + khalf16;
        const __bf16* xrow = xb + ((size_t)t * BATCH + arow) * IN_SZ + khalf16;

        auto mac = [&](v16bf a, int kt) {
            const __bf16* bp = lw + (((size_t)(kt * 4) * 32 + lane) << 4);
            v16bf b0 = *(const v16bf*)(bp);
            v16bf b1 = *(const v16bf*)(bp + 512);
            v16bf b2 = *(const v16bf*)(bp + 1024);
            v16bf b3 = *(const v16bf*)(bp + 1536);
            accF = __builtin_amdgcn_wmma_f32_16x16x32_bf16(false, a, false, b0, (short)0, accF, false, false);
            accI = __builtin_amdgcn_wmma_f32_16x16x32_bf16(false, a, false, b1, (short)0, accI, false, false);
            accC = __builtin_amdgcn_wmma_f32_16x16x32_bf16(false, a, false, b2, (short)0, accC, false, false);
            accO = __builtin_amdgcn_wmma_f32_16x16x32_bf16(false, a, false, b3, (short)0, accO, false, false);
        };

        #pragma unroll 4
        for (int kt = 0; kt < 32; ++kt)                    // K over h_{t-1}
            mac(*(const v16bf*)(hrow + kt * 32), kt);
        #pragma unroll 4
        for (int kt = 32; kt < KTILES; ++kt)               // K over x_t
            mac(*(const v16bf*)(xrow + (kt - 32) * 32), kt);

        // fused LSTM cell update (C/D layout: lanes 0-15 -> M=j, 16-31 -> M=j+8)
        #pragma unroll
        for (int j = 0; j < 8; ++j) {
            float fg = sig_f(accF[j] + biasF);
            float ig = sig_f(accI[j] + biasI);
            float gg = tanh_f(accC[j] + biasC);
            float og = sig_f(accO[j] + biasO);
            float cn = fg * c[j] + ig * gg;
            c[j] = cn;
            float h = og * tanh_f(cn);
            int b = mt * 16 + j + ((lane >> 4) << 3);
            hw[(size_t)b * HID + col] = (__bf16)h;
            out[((size_t)t * BATCH + b) * HID + col] = h;
        }

        // grid-wide barrier: release h_t, wait for all 64 workgroups
        __threadfence();
        __syncthreads();
        if (tid == 0) {
            __hip_atomic_fetch_add(cnt, 1u, __ATOMIC_RELEASE, __HIP_MEMORY_SCOPE_AGENT);
            const unsigned target = (unsigned)(t + 1) * gridDim.x;
            while (__hip_atomic_load(cnt, __ATOMIC_ACQUIRE, __HIP_MEMORY_SCOPE_AGENT) < target)
                __builtin_amdgcn_s_sleep(2);
        }
        __syncthreads();
        __threadfence();
    }
}

// ---- launch ---------------------------------------------------------------

extern "C" void kernel_launch(void* const* d_in, const int* in_sizes, int n_in,
                              void* d_out, int out_size, void* d_ws, size_t ws_size,
                              hipStream_t stream) {
    (void)in_sizes; (void)n_in; (void)out_size; (void)ws_size;
    const float* x  = (const float*)d_in[0];
    const float* Wf = (const float*)d_in[1];
    const float* bf = (const float*)d_in[2];
    const float* Wi = (const float*)d_in[3];
    const float* bi = (const float*)d_in[4];
    const float* Wc = (const float*)d_in[5];
    const float* bc = (const float*)d_in[6];
    const float* Wo = (const float*)d_in[7];
    const float* bo = (const float*)d_in[8];

    char* ws = (char*)d_ws;
    const size_t off_x  = 0;                                   // 33,554,432 B
    const size_t off_w  = off_x + (size_t)T_STEPS * BATCH * IN_SZ * 2;
    const size_t off_h0 = off_w + (size_t)4 * HID * HX * 2;    // +12,582,912
    const size_t off_h1 = off_h0 + (size_t)BATCH * HID * 2;
    const size_t off_c  = off_h1 + (size_t)BATCH * HID * 2;

    unsigned short* xbf   = (unsigned short*)(ws + off_x);
    unsigned short* wpack = (unsigned short*)(ws + off_w);
    unsigned short* h0    = (unsigned short*)(ws + off_h0);
    unsigned short* h1    = (unsigned short*)(ws + off_h1);
    unsigned*       cnt   = (unsigned*)(ws + off_c);

    lstm_cvt_x<<<(T_STEPS * BATCH * IN_SZ) / (256 * 4), 256, 0, stream>>>(x, xbf);
    lstm_pack_w<<<(NWG * KTILES * 4 * 32 + 255) / 256, 256, 0, stream>>>(Wf, Wi, Wc, Wo, wpack);
    lstm_init<<<(BATCH * HID + 255) / 256, 256, 0, stream>>>(h0, cnt);

    (void)hipFuncSetAttribute((const void*)lstm_main,
                              hipFuncAttributeMaxDynamicSharedMemorySize, WG_WBYTES);
    lstm_main<<<NWG, 128, WG_WBYTES, stream>>>(
        wpack, (const __bf16*)xbf, (__bf16*)h0, (__bf16*)h1, cnt,
        bf, bi, bc, bo, (float*)d_out);
}
